// NavigationNet_11708080849553
// MI455X (gfx1250) — matrix operand order repaired and checked by reading, not compiled, not run on hardware
//
#include <hip/hip_runtime.h>
#include <hip/hip_bf16.h>

typedef __attribute__((ext_vector_type(16))) _Float16 v16h;
typedef __attribute__((ext_vector_type(8)))  _Float16 v8h;
typedef __attribute__((ext_vector_type(8)))  float    v8f;

#define BS      1024
#define OBS_LEN 8
#define TS      64
#define HID     64

// ---- workspace layout ----
// bytes 0..1023      : bcomb_o (256 f32) = bih_o + bhh_o
// bytes 1024..2047   : bcomb_t (256 f32) = bih_t + bhh_t
// bytes 2048..       : f16 weights (offsets below in halves)
#define WS_F16_BYTES 2048
#define H_WHH_O 0          // 256*64
#define H_WHH_T 16384      // 256*64
#define H_WOF   32768      // 64*64
#define H_WTF   36864      // 64*64
#define H_WMIX  40960      // 64*128
#define H_W1    49152      // 32*64
#define H_W2    51200      // 32*32
#define H_TOTAL 52224

// ---------------- prep: weight f32->f16, fold biases, zero c_out ----------------
__global__ void prep_kernel(const float* __restrict__ Whh_o, const float* __restrict__ Whh_t,
                            const float* __restrict__ Wof,  const float* __restrict__ Wtf,
                            const float* __restrict__ Wmix, const float* __restrict__ W1,
                            const float* __restrict__ W2,
                            const float* __restrict__ bih_o, const float* __restrict__ bhh_o,
                            const float* __restrict__ bih_t, const float* __restrict__ bhh_t,
                            float* __restrict__ ws, float* __restrict__ out)
{
    _Float16* wh = (_Float16*)((char*)ws + WS_F16_BYTES);
    int tid = blockIdx.x * blockDim.x + threadIdx.x;
    int stride = gridDim.x * blockDim.x;
    for (int i = tid; i < 16384; i += stride) {
        wh[H_WHH_O + i] = (_Float16)Whh_o[i];
        wh[H_WHH_T + i] = (_Float16)Whh_t[i];
    }
    for (int i = tid; i < 4096; i += stride) {
        wh[H_WOF + i] = (_Float16)Wof[i];
        wh[H_WTF + i] = (_Float16)Wtf[i];
    }
    for (int i = tid; i < 8192; i += stride) wh[H_WMIX + i] = (_Float16)Wmix[i];
    for (int i = tid; i < 2048; i += stride) wh[H_W1 + i] = (_Float16)W1[i];
    for (int i = tid; i < 1024; i += stride) wh[H_W2 + i] = (_Float16)W2[i];
    for (int i = tid; i < 256; i += stride) {
        ws[i]       = bih_o[i] + bhh_o[i];
        ws[256 + i] = bih_t[i] + bhh_t[i];
    }
    // c_out = zeros(1024, 64), appended after y in d_out
    for (int i = tid; i < BS * TS; i += stride) out[BS * TS * 2 + i] = 0.f;
}

// ---------------- WMMA helpers ----------------
__device__ __forceinline__ v8f wmma_f16(v16h a, v16h b, v8f c) {
    return __builtin_amdgcn_wmma_f32_16x16x32_f16(false, a, false, b, (short)0, c, false, false);
}

// A fragment (16-bit A layout): lane = m + 16*hi holds K[k0 + hi*16 .. +15] (contiguous, matches B).
// Source: LDS, row-major 16 x rowStride halves.
__device__ __forceinline__ v16h loadA(const _Float16* h, int rowStride, int kbase, int m, int hi) {
    union { v16h v; v8h p[2]; } u;
    u.p[0] = *(const v8h*)(h + m * rowStride + kbase + hi * 16);
    u.p[1] = *(const v8h*)(h + m * rowStride + kbase + hi * 16 + 8);
    return u.v;
}

// B fragment (B layout): lane = n + 16*hi holds K[kk*32 + hi*16 ..+15] of column tile*16+n.
// Source: global f16 weight, row-major (N_total x K) -> 32 contiguous bytes per lane.
__device__ __forceinline__ v16h loadB(const _Float16* W, int K, int tile, int kk, int n, int hi) {
    return *(const v16h*)(W + (tile * 16 + n) * K + kk * 32 + hi * 16);
}

// Branch-free fast activations: pure v_exp_f32 / v_rcp_f32, no EXEC divergence.
// tanh limits via IEEE inf/0 propagation: exp(2x)->inf => 1-0 = 1 ; exp(2x)->0 => 1-2 = -1.
__device__ __forceinline__ float sigm(float x) {
    return __builtin_amdgcn_rcpf(1.f + __expf(-x));
}
__device__ __forceinline__ float tanh_fast(float x) {
    return 1.f - 2.f * __builtin_amdgcn_rcpf(1.f + __expf(2.f * x));
}

// ---------------- one LSTM cell step for a 16-row batch tile ----------------
// gates g = [x@Wih.T] + h@Whh.T + bcomb ; i,f,g,o split by 64 columns.
// c kept in VGPRs in WMMA C layout: c[t][r] = element (m = mBase+r, hidcol = t*16+n).
template<bool HAS_X>
__device__ __forceinline__ void lstm_cell(const _Float16* __restrict__ Whh,
                                          const float* __restrict__ bcomb,
                                          const float* __restrict__ Wih,
                                          const float* xp, int xstride,
                                          _Float16* hbuf, v8f* c,
                                          int n, int hi, int mBase)
{
    v16h a0 = loadA(hbuf, HID, 0, n, hi);
    v16h a1 = loadA(hbuf, HID, 32, n, hi);
#pragma unroll
    for (int t = 0; t < 4; ++t) {
        v8f g[4];
#pragma unroll
        for (int gt = 0; gt < 4; ++gt) {
            int col = gt * 64 + t * 16 + n;
            float base = bcomb[col];
            if (HAS_X) {
                float w0 = Wih[col * 2], w1 = Wih[col * 2 + 1];
#pragma unroll
                for (int r = 0; r < 8; ++r)
                    g[gt][r] = base + xp[(mBase + r) * xstride] * w0
                                    + xp[(mBase + r) * xstride + 1] * w1;
            } else {
#pragma unroll
                for (int r = 0; r < 8; ++r) g[gt][r] = base;
            }
            g[gt] = wmma_f16(a0, loadB(Whh, HID, gt * 4 + t, 0, n, hi), g[gt]);
            g[gt] = wmma_f16(a1, loadB(Whh, HID, gt * 4 + t, 1, n, hi), g[gt]);
        }
#pragma unroll
        for (int r = 0; r < 8; ++r) {
            float iv = sigm(g[0][r]);
            float fv = sigm(g[1][r]);
            float gv = tanh_fast(g[2][r]);
            float ov = sigm(g[3][r]);
            float cn = fv * c[t][r] + iv * gv;
            c[t][r] = cn;
            hbuf[(mBase + r) * HID + t * 16 + n] = (_Float16)(ov * tanh_fast(cn));
        }
    }
}

// ---------------- generic 16xK @ K x (NT*16) GEMM + bias (+lrelu), out to LDS ----------------
template<int K, int NT, bool ACT, bool F16OUT>
__device__ __forceinline__ void gemm16(const _Float16* __restrict__ W,
                                       const _Float16* A, int aStride,
                                       const float* __restrict__ bias,
                                       _Float16* outH, float* outF, int outStride, int outCol,
                                       int n, int hi, int mBase)
{
    v16h af[K / 32];
#pragma unroll
    for (int kk = 0; kk < K / 32; ++kk) af[kk] = loadA(A, aStride, kk * 32, n, hi);
#pragma unroll
    for (int t = 0; t < NT; ++t) {
        v8f acc;
        float b = bias[t * 16 + n];
#pragma unroll
        for (int r = 0; r < 8; ++r) acc[r] = b;
#pragma unroll
        for (int kk = 0; kk < K / 32; ++kk)
            acc = wmma_f16(af[kk], loadB(W, K, t, kk, n, hi), acc);
#pragma unroll
        for (int r = 0; r < 8; ++r) {
            float v = acc[r];
            if (ACT) v = (v >= 0.f) ? v : 0.1f * v;
            int idx = (mBase + r) * outStride + outCol + t * 16 + n;
            if (F16OUT) outH[idx] = (_Float16)v; else outF[idx] = v;
        }
    }
}

// ---------------- main rollout: 1 wave = 16 batch rows, loops all 72 timesteps ----------------
__global__ void __launch_bounds__(32)
nav_kernel(const float* __restrict__ obsv, const float* __restrict__ Wih_o,
           const float* __restrict__ btf, const float* __restrict__ bof,
           const float* __restrict__ bmix, const float* __restrict__ b1,
           const float* __restrict__ b2, const float* __restrict__ W3,
           const float* __restrict__ b3,
           const float* __restrict__ ws, float* __restrict__ out)
{
    __shared__ alignas(32) _Float16 hO[16 * HID];
    __shared__ alignas(32) _Float16 hT[16 * HID];
    __shared__ alignas(32) _Float16 cc[16 * 128];   // concat [ts, os]
    __shared__ alignas(32) _Float16 mb[16 * HID];   // mix output
    __shared__ alignas(32) _Float16 a1b[16 * 32];
    __shared__ alignas(32) float    a2f[16 * 32];
    __shared__ float buf[16 * 6];                   // ring of last 3 y per row
    __shared__ float xv[16 * 2];
    __shared__ float w3s[64];                       // W3 (2x32) staged once
    __shared__ float b3s[2];

    const _Float16* wh   = (const _Float16*)((const char*)ws + WS_F16_BYTES);
    const _Float16* whhO = wh + H_WHH_O;
    const _Float16* whhT = wh + H_WHH_T;
    const _Float16* wof  = wh + H_WOF;
    const _Float16* wtf  = wh + H_WTF;
    const _Float16* wmix = wh + H_WMIX;
    const _Float16* w1   = wh + H_W1;
    const _Float16* w2   = wh + H_W2;
    const float* bcO = ws;
    const float* bcT = ws + 256;

    const int l = threadIdx.x;
    const int n = l & 15;
    const int hi = l >> 4;
    const int mBase = hi * 8;
    const int b0 = blockIdx.x * 16;

    // warm the f16 weight block into cache (global_prefetch)
    for (int off = l * 128; off < H_TOTAL * 2; off += 32 * 128)
        __builtin_prefetch((const char*)wh + off, 0, 3);

    // stage the tiny head weights
    w3s[l] = W3[l];
    w3s[32 + l] = W3[32 + l];
    if (l < 2) b3s[l] = b3[l];

    v8f cO[4], cT[4];
#pragma unroll
    for (int t = 0; t < 4; ++t)
#pragma unroll
        for (int r = 0; r < 8; ++r) { cO[t][r] = 0.f; cT[t][r] = 0.f; }
    for (int i = l; i < 16 * HID; i += 32) { hO[i] = (_Float16)0.f; hT[i] = (_Float16)0.f; }
    __syncthreads();

    // ---- observation encoder: 8 steps of the obsv LSTM ----
    for (int t = 0; t < OBS_LEN; ++t) {
        { int row = l >> 1, comp = l & 1;
          xv[row * 2 + comp] = obsv[((b0 + row) * OBS_LEN + t) * 2 + comp]; }
        __syncthreads();
        lstm_cell<true>(whhO, bcO, Wih_o, xv, 2, hO, cO, n, hi, mBase);
        __syncthreads();
    }

    // ---- first teom cell (its input is identically zero in the reference) ----
    lstm_cell<false>(whhT, bcT, nullptr, nullptr, 0, hT, cT, n, hi, mBase);
    __syncthreads();

    // ---- buf = obsv[:, 5:8, :] ----
    { int row = l >> 1, comp = l & 1;
#pragma unroll
      for (int j = 0; j < 3; ++j)
          buf[row * 6 + j * 2 + comp] = obsv[((b0 + row) * OBS_LEN + 5 + j) * 2 + comp];
    }
    __syncthreads();

    // ---- 64 decode steps ----
    for (int step = 0; step < TS; ++step) {
        if (step > 0) {
            lstm_cell<false>(whhT, bcT, nullptr, nullptr, 0, hT, cT, n, hi, mBase);
            lstm_cell<true>(whhO, bcO, Wih_o, buf + 4, 6, hO, cO, n, hi, mBase);
            __syncthreads();
        }
        // ts = hT@Wtf.T+btf -> cc[:,0:64] ; os = hO@Wof.T+bof -> cc[:,64:128]
        gemm16<64, 4, false, true >(wtf, hT, HID, btf, cc, nullptr, 128, 0,  n, hi, mBase);
        gemm16<64, 4, false, true >(wof, hO, HID, bof, cc, nullptr, 128, 64, n, hi, mBase);
        __syncthreads();
        // m = cc@Wmix.T+bmix
        gemm16<128, 4, false, true>(wmix, cc, 128, bmix, mb, nullptr, 64, 0, n, hi, mBase);
        __syncthreads();
        // a1 = lrelu(m@W1.T+b1)
        gemm16<64, 2, true, true  >(w1, mb, 64, b1, a1b, nullptr, 32, 0, n, hi, mBase);
        __syncthreads();
        // a2 = lrelu(a1@W2.T+b2) (f32 out)
        gemm16<32, 2, true, false >(w2, a1b, 32, b2, nullptr, a2f, 32, 0, n, hi, mBase);
        __syncthreads();
        // head: o = a2@W3.T+b3 ; y = o + buf[-1] + (buf[-1]-buf[0])*0.5 ; shift buf; store y
        { int row = n, d = hi;  // 16 rows x 2 output dims = 32 lanes
          float acc = b3s[d];
#pragma unroll
          for (int k = 0; k < 32; ++k) acc += a2f[row * 32 + k] * w3s[d * 32 + k];
          float f0 = buf[row * 6 + 0 + d];
          float f1 = buf[row * 6 + 2 + d];
          float f2 = buf[row * 6 + 4 + d];
          float y  = acc + f2 + (f2 - f0) * 0.5f;
          buf[row * 6 + 0 + d] = f1;
          buf[row * 6 + 2 + d] = f2;
          buf[row * 6 + 4 + d] = y;
          out[((b0 + row) * TS + step) * 2 + d] = y;
        }
        __syncthreads();
    }
}

extern "C" void kernel_launch(void* const* d_in, const int* in_sizes, int n_in,
                              void* d_out, int out_size, void* d_ws, size_t ws_size,
                              hipStream_t stream)
{
    const float* obsv  = (const float*)d_in[0];
    // d_in[1] (teom, 256 MB) is intentionally unused: reference feeds zeros to the teom LSTM.
    const float* Wih_o = (const float*)d_in[2];
    const float* Whh_o = (const float*)d_in[3];
    const float* bih_o = (const float*)d_in[4];
    const float* bhh_o = (const float*)d_in[5];
    const float* Whh_t = (const float*)d_in[7];
    const float* bih_t = (const float*)d_in[8];
    const float* bhh_t = (const float*)d_in[9];
    const float* Wof   = (const float*)d_in[10];
    const float* bof   = (const float*)d_in[11];
    const float* Wtf   = (const float*)d_in[12];
    const float* btf   = (const float*)d_in[13];
    const float* Wmix  = (const float*)d_in[14];
    const float* bmix  = (const float*)d_in[15];
    const float* W1    = (const float*)d_in[16];
    const float* b1    = (const float*)d_in[17];
    const float* W2    = (const float*)d_in[18];
    const float* b2    = (const float*)d_in[19];
    const float* W3    = (const float*)d_in[20];
    const float* b3    = (const float*)d_in[21];
    float* out = (float*)d_out;
    float* ws  = (float*)d_ws;

    prep_kernel<<<128, 256, 0, stream>>>(Whh_o, Whh_t, Wof, Wtf, Wmix, W1, W2,
                                         bih_o, bhh_o, bih_t, bhh_t, ws, out);
    nav_kernel<<<BS / 16, 32, 0, stream>>>(obsv, Wih_o, btf, bof, bmix, b1, b2, W3, b3,
                                           ws, out);
}
